// SSM_57123065037327
// MI455X (gfx1250) — matrix operand order repaired and checked by reading, not compiled, not run on hardware
//
#include <hip/hip_runtime.h>
#include <hip/hip_bf16.h>
#include <math.h>

// ---------------- problem constants ----------------
constexpr int EMB   = 2048;
constexpr int HID   = 4096;
constexpr int STATE = 128;
constexpr int NH    = 64;
constexpr int KCONV = 4;
constexpr int CS    = 256;
constexpr int DH    = HID / NH;              // 64
constexpr int CONV_DIM = HID + 2 * STATE;    // 4352
constexpr int HSZ   = 2 * HID + 2 * STATE + NH; // 8512
constexpr int BSZ   = 2;
constexpr int SEQ   = 2048;
constexpr int NC    = SEQ / CS;              // 8
constexpr int TOK   = BSZ * SEQ;             // 4096
constexpr int NCHUNK = BSZ * NC;             // 16

// ---------------- WMMA helpers ----------------
typedef __attribute__((ext_vector_type(16))) __bf16 v16bf;
typedef __attribute__((ext_vector_type(8)))  __bf16 v8bf;
typedef __attribute__((ext_vector_type(4)))  __bf16 v4bf;
typedef __attribute__((ext_vector_type(2)))  __bf16 v2bf;
typedef __attribute__((ext_vector_type(8)))  float  v8f;

constexpr int GSTR = 40;  // LDS row stride in bf16 (80 B rows: 16B-aligned, bank-spread)

// A-fragment: lane holds row m = base+(lane&15); elements e: K = (e>>3)*16 + half*8 + (e&7)
__device__ __forceinline__ v16bf frag_a(const __bf16* base, int row, int lane) {
  const int hh = (lane >> 4) & 1;
  const __bf16* p = base + row * GSTR;
  v8bf lo = *(const v8bf*)(p + hh * 8);
  v8bf hi = *(const v8bf*)(p + 16 + hh * 8);
  v16bf f;
#pragma unroll
  for (int e = 0; e < 8; ++e) { f[e] = lo[e]; f[8 + e] = hi[e]; }
  return f;
}
// B-fragment: lane holds col n = base+(lane&15); elements e: K = (e>>3)*16 + half*8 + (e&7)
__device__ __forceinline__ v16bf frag_b(const __bf16* base, int row, int lane) {
  const int hh = (lane >> 4) & 1;
  const __bf16* p = base + row * GSTR + hh * 8;
  v8bf lo = *(const v8bf*)(p);
  v8bf hi = *(const v8bf*)(p + 16);
  v16bf f;
#pragma unroll
  for (int e = 0; e < 8; ++e) { f[e] = lo[e]; f[8 + e] = hi[e]; }
  return f;
}
__device__ __forceinline__ v8f wmma_bf16(v16bf a, v16bf b, v8f c) {
  return __builtin_amdgcn_wmma_f32_16x16x32_bf16(false, a, false, b, short(0), c, false, false);
}
__device__ __forceinline__ float siluf(float v) { return v / (1.f + expf(-v)); }
__device__ __forceinline__ v2bf pk2(float x, float y) {
  v2bf t; t[0] = (__bf16)x; t[1] = (__bf16)y; return t;
}

// ---- Kernels 1/9: C[M,N] = A[M,K] * B[N,K]^T, BM=128 BN=256 BK=32, dbl-buffered --
template <bool A_BF16>
__global__ void __launch_bounds__(256)
gemm_wmma_kernel(const void* __restrict__ Ap, const float* __restrict__ B,
                 float* __restrict__ C, int M, int N, int Kd) {
  __shared__ __bf16 lA[2][128 * GSTR];  // 20 KB
  __shared__ __bf16 lB[2][256 * GSTR];  // 40 KB
  const int tid = threadIdx.x, lane = tid & 31, wv = tid >> 5;
  const int lanem = lane & 15, hh = (lane >> 4) & 1;
  const int m0 = blockIdx.y * 128, n0 = blockIdx.x * 256;
  const int wm = wv >> 2, wn = wv & 3;  // wave: 64 rows x 64 cols
  v8f acc[4][4] = {};

  auto stage = [&](int buf, int k0) {
#pragma unroll
    for (int it = 0; it < 4; ++it) {              // A: 128x32 = 1024 float4 units
      const int idx = tid + it * 256;
      const int row = idx >> 3, c4 = (idx & 7) * 4;
      v4bf a4;
      if constexpr (A_BF16) {
        const __bf16* Af = (const __bf16*)Ap;
        a4 = *(const v4bf*)(Af + (size_t)(m0 + row) * Kd + k0 + c4);
      } else {
        const float* Af = (const float*)Ap;
        const float4 f = *(const float4*)(Af + (size_t)(m0 + row) * Kd + k0 + c4);
        a4[0] = (__bf16)f.x; a4[1] = (__bf16)f.y; a4[2] = (__bf16)f.z; a4[3] = (__bf16)f.w;
      }
      *(v4bf*)(&lA[buf][row * GSTR + c4]) = a4;
    }
#pragma unroll
    for (int it = 0; it < 8; ++it) {              // B: 256x32 = 2048 float4 units
      const int idx = tid + it * 256;
      const int row = idx >> 3, c4 = (idx & 7) * 4;
      const int gn = n0 + row;
      v4bf b4;
      if (gn < N) {
        const float4 f = *(const float4*)(B + (size_t)gn * Kd + k0 + c4);
        b4[0] = (__bf16)f.x; b4[1] = (__bf16)f.y; b4[2] = (__bf16)f.z; b4[3] = (__bf16)f.w;
        if (it == 0 && k0 + 32 < Kd)
          __builtin_prefetch((const void*)(B + (size_t)gn * Kd + k0 + 32), 0, 3);
      } else {
        b4[0] = b4[1] = b4[2] = b4[3] = (__bf16)0.f;
      }
      *(v4bf*)(&lB[buf][row * GSTR + c4]) = b4;
    }
  };

  stage(0, 0);
  __syncthreads();
  const int nkb = Kd / 32;
  for (int kb = 0; kb < nkb; ++kb) {
    const int buf = kb & 1;
    if (kb + 1 < nkb) stage(buf ^ 1, (kb + 1) * 32);  // overlap with WMMA below
    v16bf afr[4], bfr[4];
#pragma unroll
    for (int i = 0; i < 4; ++i) afr[i] = frag_a(&lA[buf][0], wm * 64 + i * 16 + lanem, lane);
#pragma unroll
    for (int j = 0; j < 4; ++j) bfr[j] = frag_b(&lB[buf][0], wn * 64 + j * 16 + lanem, lane);
#pragma unroll
    for (int i = 0; i < 4; ++i)
#pragma unroll
      for (int j = 0; j < 4; ++j) acc[i][j] = wmma_bf16(afr[i], bfr[j], acc[i][j]);
    __syncthreads();
  }
#pragma unroll
  for (int i = 0; i < 4; ++i) {
    const int gmb = m0 + wm * 64 + i * 16 + hh * 8;
#pragma unroll
    for (int j = 0; j < 4; ++j) {
      const int gn = n0 + wn * 64 + j * 16 + lanem;
      if (gn < N) {
#pragma unroll
        for (int r = 0; r < 8; ++r) C[(size_t)(gmb + r) * N + gn] = acc[i][j][r];
      }
    }
  }
}

// ---------------- Kernel 2: depthwise causal conv (K=4) + silu -------------------
__global__ void conv_silu_kernel(const float* __restrict__ z, const float* __restrict__ w,
                                 float* __restrict__ xc) {
  const int idx = blockIdx.x * blockDim.x + threadIdx.x;
  if (idx >= TOK * CONV_DIM) return;
  const int c = idx % CONV_DIM;
  const int t = idx / CONV_DIM;
  const int s = t % SEQ, b = t / SEQ;
  float acc = 0.f;
#pragma unroll
  for (int k = 0; k < KCONV; ++k) {
    const int sp = s + k - (KCONV - 1);
    if (sp >= 0) acc += z[((size_t)(b * SEQ + sp)) * HSZ + HID + c] * w[c * KCONV + k];
  }
  xc[idx] = siluf(acc);
}

// ---------------- Kernel 3: per-(chunk,head) cumsum of dt*A ----------------------
__global__ void cum_kernel(const float* __restrict__ z, const float* __restrict__ A_log,
                           float* __restrict__ cum, float* __restrict__ dtr) {
  const int h = threadIdx.x;          // 0..63
  const int chunk = blockIdx.x;       // 0..15 (b-major)
  const float Ah = -expf(A_log[h]);
  float c = 0.f;
  for (int i = 0; i < CS; ++i) {
    const size_t tg = (size_t)chunk * CS + i;
    const float dt = z[tg * HSZ + HID + CONV_DIM + h];
    c += dt * Ah;
    cum[tg * NH + h] = c;
    dtr[tg * NH + h] = dt;
  }
}

// ---------------- Kernel 4: CB[chunk,i,j] = sum_n C[i,n]*B[j,n] ------------------
__global__ void __launch_bounds__(256)
cb_kernel(const float* __restrict__ xc, float* __restrict__ CB) {
  __shared__ __bf16 lA[64 * GSTR];
  __shared__ __bf16 lB[256 * GSTR];
  const int chunk = blockIdx.y;
  const int i0 = blockIdx.x * 64;
  const int tid = threadIdx.x, lane = tid & 31, wv = tid >> 5;
  const int lanem = lane & 15, hh = (lane >> 4) & 1;
  const int wm = wv & 3, wn = wv >> 2;  // wave: 16 rows x 128 cols (8 n-tiles)
  v8f acc[8] = {};
  for (int kb = 0; kb < 4; ++kb) {
    const int nb = kb * 32;
#pragma unroll
    for (int it = 0; it < 4; ++it) {    // A: 64x32 = 1024 pairs
      const int idx = tid + it * 256;
      const int r = idx >> 4, c2 = (idx & 15) * 2;
      const float2 f = *(const float2*)(xc + (size_t)(chunk * CS + i0 + r) * CONV_DIM +
                                        HID + STATE + nb + c2);
      *(v2bf*)(&lA[r * GSTR + c2]) = pk2(f.x, f.y);
    }
#pragma unroll
    for (int it = 0; it < 16; ++it) {   // B: 256x32 = 4096 pairs
      const int idx = tid + it * 256;
      const int r = idx >> 4, c2 = (idx & 15) * 2;
      const float2 f = *(const float2*)(xc + (size_t)(chunk * CS + r) * CONV_DIM +
                                        HID + nb + c2);
      *(v2bf*)(&lB[r * GSTR + c2]) = pk2(f.x, f.y);
    }
    __syncthreads();
    v16bf afr = frag_a(lA, wm * 16 + lanem, lane);
    v16bf bfr[8];
#pragma unroll
    for (int jj = 0; jj < 8; ++jj) bfr[jj] = frag_b(lB, wn * 128 + jj * 16 + lanem, lane);
#pragma unroll
    for (int jj = 0; jj < 8; ++jj) acc[jj] = wmma_bf16(afr, bfr[jj], acc[jj]);
    __syncthreads();
  }
#pragma unroll
  for (int jj = 0; jj < 8; ++jj)
#pragma unroll
    for (int r = 0; r < 8; ++r)
      CB[(size_t)chunk * CS * CS + (size_t)(i0 + wm * 16 + hh * 8 + r) * CS +
         (wn * 128 + jj * 16 + lanem)] = acc[jj][r];
}

// ---------------- Kernel 5: per-chunk states[d,n] = sum_j w_j x[j,d] B[j,n] ------
__global__ void __launch_bounds__(256)
states_kernel(const float* __restrict__ xc, const float* __restrict__ cum,
              const float* __restrict__ dtr, float* __restrict__ states) {
  __shared__ __bf16 lA[64 * GSTR];    // [d][j]
  __shared__ __bf16 lB[128 * GSTR];   // [n][j]
  const int p = blockIdx.x, chunk = blockIdx.y;
  const int tid = threadIdx.x, lane = tid & 31, wv = tid >> 5;
  const int lanem = lane & 15, hh = (lane >> 4) & 1;
  const int wm = wv & 3, wn = wv >> 2;
  const float clast = cum[((size_t)chunk * CS + (CS - 1)) * NH + p];
  v8f acc[4] = {};
  for (int kb = 0; kb < 8; ++kb) {
    const int j0 = kb * 32;
#pragma unroll
    for (int it = 0; it < 4; ++it) {    // A: 32j x 64d, pairs along d (transposed write)
      const int idx = tid + it * 256;
      const int j = idx >> 5, d2 = (idx & 31) * 2;
      const size_t tg = (size_t)chunk * CS + j0 + j;
      const float wj = expf(clast - cum[tg * NH + p]) * dtr[tg * NH + p];
      const float2 f = *(const float2*)(xc + tg * CONV_DIM + p * DH + d2);
      lA[d2 * GSTR + j]       = (__bf16)(wj * f.x);
      lA[(d2 + 1) * GSTR + j] = (__bf16)(wj * f.y);
    }
#pragma unroll
    for (int it = 0; it < 8; ++it) {    // B: 32j x 128n, pairs along n (transposed write)
      const int idx = tid + it * 256;
      const int j = idx >> 6, n2 = (idx & 63) * 2;
      const size_t tg = (size_t)chunk * CS + j0 + j;
      const float2 f = *(const float2*)(xc + tg * CONV_DIM + HID + n2);
      lB[n2 * GSTR + j]       = (__bf16)f.x;
      lB[(n2 + 1) * GSTR + j] = (__bf16)f.y;
    }
    __syncthreads();
    v16bf afr = frag_a(lA, wm * 16 + lanem, lane);
    v16bf bfr[4];
#pragma unroll
    for (int jj = 0; jj < 4; ++jj) bfr[jj] = frag_b(lB, wn * 64 + jj * 16 + lanem, lane);
#pragma unroll
    for (int jj = 0; jj < 4; ++jj) acc[jj] = wmma_bf16(afr, bfr[jj], acc[jj]);
    __syncthreads();
  }
#pragma unroll
  for (int jj = 0; jj < 4; ++jj)
#pragma unroll
    for (int r = 0; r < 8; ++r) {
      const int d = wm * 16 + hh * 8 + r;
      const int n = wn * 64 + jj * 16 + lanem;
      states[(((size_t)chunk * NH + p) * DH + d) * STATE + n] = acc[jj][r];
    }
}

// ---------------- Kernel 6: 8-step inter-chunk state scan ------------------------
__global__ void scan_kernel(const float* __restrict__ states, const float* __restrict__ cum,
                            float* __restrict__ S_in) {
  const int idx = blockIdx.x * blockDim.x + threadIdx.x;
  if (idx >= BSZ * NH * DH * STATE) return;
  const int n = idx % STATE;
  int r = idx / STATE;
  const int d = r % DH; r /= DH;
  const int p = r % NH;
  const int b = r / NH;
  float S = 0.f;
  for (int zc = 0; zc < NC; ++zc) {
    const int chunk = b * NC + zc;
    const size_t off = (((size_t)chunk * NH + p) * DH + d) * STATE + n;
    S_in[off] = S;
    const float cd = expf(cum[((size_t)chunk * CS + (CS - 1)) * NH + p]);
    S = cd * S + states[off];
  }
}

// ---------------- Kernel 7: Y = (CB*L*dt)@x + exp(cum)*C@S_in^T + D*x ------------
__global__ void __launch_bounds__(256)
y_kernel(const float* __restrict__ xc, const float* __restrict__ cum,
         const float* __restrict__ dtr, const float* __restrict__ CB,
         const float* __restrict__ S_in, const float* __restrict__ Dv,
         float* __restrict__ Y) {
  __shared__ __bf16 lA[256 * GSTR];
  __shared__ __bf16 lB[64 * GSTR];
  const int p = blockIdx.x, chunk = blockIdx.y;
  const int tid = threadIdx.x, lane = tid & 31, wv = tid >> 5;
  const int lanem = lane & 15, hh = (lane >> 4) & 1;
  v8f acc[2][4] = {};   // wave covers rows wv*32..+31, all 64 cols
  // phase 1: intra-chunk masked matmul, K = CS
  for (int kb = 0; kb < 8; ++kb) {
    const int j0 = kb * 32;
#pragma unroll
    for (int it = 0; it < 16; ++it) {   // M_p: 256i x 32j, pairs along j
      const int idx = tid + it * 256;
      const int i = idx >> 4, j2 = (idx & 15) * 2;
      const float ci = cum[((size_t)chunk * CS + i) * NH + p];
      const float2 cb2 = *(const float2*)(CB + (size_t)chunk * CS * CS + (size_t)i * CS + j0 + j2);
      float m0v = 0.f, m1v = 0.f;
      const int jg0 = j0 + j2, jg1 = j0 + j2 + 1;
      if (jg0 <= i)
        m0v = cb2.x * expf(ci - cum[((size_t)chunk * CS + jg0) * NH + p]) *
              dtr[((size_t)chunk * CS + jg0) * NH + p];
      if (jg1 <= i)
        m1v = cb2.y * expf(ci - cum[((size_t)chunk * CS + jg1) * NH + p]) *
              dtr[((size_t)chunk * CS + jg1) * NH + p];
      *(v2bf*)(&lA[i * GSTR + j2]) = pk2(m0v, m1v);
    }
#pragma unroll
    for (int it = 0; it < 4; ++it) {    // xh: 32j x 64d, pairs along d (transposed)
      const int idx = tid + it * 256;
      const int j = idx >> 5, d2 = (idx & 31) * 2;
      const size_t tg = (size_t)chunk * CS + j0 + j;
      const float2 f = *(const float2*)(xc + tg * CONV_DIM + p * DH + d2);
      lB[d2 * GSTR + j]       = (__bf16)f.x;
      lB[(d2 + 1) * GSTR + j] = (__bf16)f.y;
    }
    __syncthreads();
    v16bf afr[2], bfr[4];
#pragma unroll
    for (int it = 0; it < 2; ++it) afr[it] = frag_a(lA, (wv * 2 + it) * 16 + lanem, lane);
#pragma unroll
    for (int jt = 0; jt < 4; ++jt) bfr[jt] = frag_b(lB, jt * 16 + lanem, lane);
#pragma unroll
    for (int it = 0; it < 2; ++it)
#pragma unroll
      for (int jt = 0; jt < 4; ++jt) acc[it][jt] = wmma_bf16(afr[it], bfr[jt], acc[it][jt]);
    __syncthreads();
  }
  // phase 2: inter-chunk contribution, K = STATE
  for (int kb = 0; kb < 4; ++kb) {
    const int n0 = kb * 32;
#pragma unroll
    for (int it = 0; it < 16; ++it) {   // exp(cum)*C: 256i x 32n, pairs along n
      const int idx = tid + it * 256;
      const int i = idx >> 4, n2 = (idx & 15) * 2;
      const size_t tg = (size_t)chunk * CS + i;
      const float e = expf(cum[tg * NH + p]);
      const float2 f = *(const float2*)(xc + tg * CONV_DIM + HID + STATE + n0 + n2);
      *(v2bf*)(&lA[i * GSTR + n2]) = pk2(e * f.x, e * f.y);
    }
#pragma unroll
    for (int it = 0; it < 4; ++it) {    // S_in: 64d x 32n, pairs along n
      const int idx = tid + it * 256;
      const int d = idx >> 4, n2 = (idx & 15) * 2;
      const float2 f = *(const float2*)(S_in + (((size_t)chunk * NH + p) * DH + d) * STATE +
                                        n0 + n2);
      *(v2bf*)(&lB[d * GSTR + n2]) = pk2(f.x, f.y);
    }
    __syncthreads();
    v16bf afr[2], bfr[4];
#pragma unroll
    for (int it = 0; it < 2; ++it) afr[it] = frag_a(lA, (wv * 2 + it) * 16 + lanem, lane);
#pragma unroll
    for (int jt = 0; jt < 4; ++jt) bfr[jt] = frag_b(lB, jt * 16 + lanem, lane);
#pragma unroll
    for (int it = 0; it < 2; ++it)
#pragma unroll
      for (int jt = 0; jt < 4; ++jt) acc[it][jt] = wmma_bf16(afr[it], bfr[jt], acc[it][jt]);
    __syncthreads();
  }
  // epilogue: + D[p]*xh, store out2
  const float Dp = Dv[p];
#pragma unroll
  for (int it = 0; it < 2; ++it)
#pragma unroll
    for (int jt = 0; jt < 4; ++jt)
#pragma unroll
      for (int r = 0; r < 8; ++r) {
        const int i = wv * 32 + it * 16 + hh * 8 + r;
        const int d = jt * 16 + lanem;
        const size_t tg = (size_t)chunk * CS + i;
        Y[tg * HID + p * DH + d] =
            acc[it][jt][r] + Dp * xc[tg * CONV_DIM + p * DH + d];
      }
}

// ---------------- Kernel 8: gate (silu(out1)*out2), RMS-norm, emit bf16 ----------
__global__ void __launch_bounds__(256)
gatenorm_kernel(const float* __restrict__ z, const float* __restrict__ y,
                const float* __restrict__ nw, __bf16* __restrict__ out) {
  const int tok = blockIdx.x, tid = threadIdx.x;
  __shared__ float red[8];
  float vals[16];
  float ss = 0.f;
#pragma unroll
  for (int i = 0; i < 16; ++i) {
    const int c = i * 256 + tid;
    const float o1 = z[(size_t)tok * HSZ + c];
    const float g = siluf(o1) * y[(size_t)tok * HID + c];
    vals[i] = g;
    ss += g * g;
  }
#pragma unroll
  for (int off = 16; off >= 1; off >>= 1) ss += __shfl_xor(ss, off, 32);
  if ((tid & 31) == 0) red[tid >> 5] = ss;
  __syncthreads();
  float tot = 0.f;
#pragma unroll
  for (int w8 = 0; w8 < 8; ++w8) tot += red[w8];
  const float scale = rsqrtf(tot / (float)HID + 1e-5f);
#pragma unroll
  for (int i = 0; i < 16; ++i) {
    const int c = i * 256 + tid;
    out[(size_t)tok * HID + c] = (__bf16)(vals[i] * scale * nw[c]);
  }
}

// ---------------- host launch ----------------------------------------------------
extern "C" void kernel_launch(void* const* d_in, const int* in_sizes, int n_in,
                              void* d_out, int out_size, void* d_ws, size_t ws_size,
                              hipStream_t stream) {
  const float* x     = (const float*)d_in[0];
  const float* W_in  = (const float*)d_in[1];
  const float* W_out = (const float*)d_in[2];
  const float* convw = (const float*)d_in[3];
  const float* A_log = (const float*)d_in[4];
  const float* Dv    = (const float*)d_in[5];
  const float* nw    = (const float*)d_in[6];
  float* out = (float*)d_out;

  char* p = (char*)d_ws;
  float* z      = (float*)p;            p += (size_t)TOK * HSZ * 4;       // 139.5 MB
  float* xc     = (float*)p;            p += (size_t)TOK * CONV_DIM * 4;  //  71.3 MB
  float* cum    = (float*)p;            p += (size_t)TOK * NH * 4;
  float* dtr    = (float*)p;            p += (size_t)TOK * NH * 4;
  float* CB     = (float*)p;            p += (size_t)NCHUNK * CS * CS * 4;
  float* states = (float*)p;            p += (size_t)NCHUNK * NH * DH * STATE * 4;
  float* S_in   = (float*)p;            p += (size_t)NCHUNK * NH * DH * STATE * 4;
  float* Ybuf   = (float*)p;            p += (size_t)TOK * HID * 4;
  __bf16* nbf   = (__bf16*)p;           p += (size_t)TOK * HID * 2;

  // 1) z = x @ W_in^T  (M=4096, N=8512, K=2048)
  gemm_wmma_kernel<false><<<dim3((HSZ + 255) / 256, TOK / 128), 256, 0, stream>>>(
      x, W_in, z, TOK, HSZ, EMB);
  // 2) depthwise conv + silu
  conv_silu_kernel<<<((size_t)TOK * CONV_DIM + 255) / 256, 256, 0, stream>>>(z, convw, xc);
  // 3) cumsum of dt*A per (chunk, head)
  cum_kernel<<<NCHUNK, NH, 0, stream>>>(z, A_log, cum, dtr);
  // 4) CB = C @ B^T per chunk
  cb_kernel<<<dim3(4, NCHUNK), 256, 0, stream>>>(xc, CB);
  // 5) per-chunk state outer products
  states_kernel<<<dim3(NH, NCHUNK), 256, 0, stream>>>(xc, cum, dtr, states);
  // 6) inter-chunk scan
  scan_kernel<<<(BSZ * NH * DH * STATE + 255) / 256, 256, 0, stream>>>(states, cum, S_in);
  // 7) Y (intra + inter + D*x)
  y_kernel<<<dim3(NH, NCHUNK), 256, 0, stream>>>(xc, cum, dtr, CB, S_in, Dv, Ybuf);
  // 8) gate + RMS norm -> bf16
  gatenorm_kernel<<<TOK, 256, 0, stream>>>(z, Ybuf, nw, nbf);
  // 9) out = normed @ W_out^T  (M=4096, N=2048, K=4096)
  gemm_wmma_kernel<true><<<dim3(EMB / 256, TOK / 128), 256, 0, stream>>>(
      nbf, W_out, out, TOK, EMB, HID);
}
